// MyDecoderLayer_68959994904757
// MI455X (gfx1250) — compile-verified, matrix operand order checked by reading, not run on hardware
//
#include <hip/hip_runtime.h>
#include <hip/hip_bf16.h>
#include <math.h>

// ---------------------------------------------------------------------------
// Types for CDNA5 WMMA (wave32)
// ---------------------------------------------------------------------------
typedef __bf16 v16bf __attribute__((ext_vector_type(16)));
typedef __bf16 v8bf  __attribute__((ext_vector_type(8)));
typedef __bf16 v4bf  __attribute__((ext_vector_type(4)));
typedef float  v8f   __attribute__((ext_vector_type(8)));
typedef float  v4f   __attribute__((ext_vector_type(4)));

// Native f32 -> bf16 truncation (lowers to packed v_cvt on gfx1250).
__device__ __forceinline__ __bf16 f2bf(float f) { return (__bf16)f; }
__device__ __forceinline__ v4bf f2bf4(v4f v) {
    return __builtin_convertvector(v, v4bf);
}

__device__ __forceinline__ float gelu_f(float x) {
    return 0.5f * x * (1.0f + erff(x * 0.70710678118654752f));
}

// Load a 16-element bf16 A/B fragment from an LDS row (row-major over K,
// 32 K values per row). lane<16 takes K {0..7,16..23}; lane>=16 takes
// K {8..15,24..31} (CDNA5 16-bit 16x32 operand striping).
__device__ __forceinline__ v16bf load_frag(const __bf16* rowp, int kb) {
    v8bf lo = *(const v8bf*)(rowp + kb);
    v8bf hi = *(const v8bf*)(rowp + kb + 16);
    v16bf r;
#pragma unroll
    for (int i = 0; i < 8; ++i) { r[i] = lo[i]; r[i + 8] = hi[i]; }
    return r;
}

// ---------------------------------------------------------------------------
// Generic WMMA GEMM:  D[M,N] = epi( A[M,K] @ B + bias + resid )
//   BL  = 0 : B stored [K,N]  (out = A @ W)
//   BL  = 1 : B stored [N,K]  (out = A @ W^T, conv1x1 weights [Co,Ci])
//   EPI : 0 none, 1 relu, 2 gelu;  HB/HR: compile-time bias/residual enables.
// Block: 128 threads (4 waves). Tile 128x64, K-step 32; each wave owns a
// 32x64 strip (2 A fragments x 4 B fragments = 8 WMMAs per K-step).
// M % 128 == 0, N % 64 == 0, K % 32 == 0 (all shapes here qualify).
// ---------------------------------------------------------------------------
template <int BL, int EPI, bool HB, bool HR>
__global__ void gemm_wmma_kernel(const float* __restrict__ A,
                                 const float* __restrict__ Bw,
                                 const float* __restrict__ bias,
                                 const float* __restrict__ resid,
                                 float* __restrict__ D,
                                 int M, int N, int K) {
    __shared__ __attribute__((aligned(16))) __bf16 As[128][40];
    __shared__ __attribute__((aligned(16))) __bf16 Bs[64][40];

    const int tid  = threadIdx.x;
    const int wave = tid >> 5;
    const int lane = tid & 31;
    const int m0 = blockIdx.x * 128;
    const int n0 = blockIdx.y * 64;

    v8f acc[2][4];
#pragma unroll
    for (int h = 0; h < 2; ++h)
#pragma unroll
        for (int t = 0; t < 4; ++t) acc[h][t] = (v8f)(0.0f);

    for (int k0 = 0; k0 < K; k0 += 32) {
        // Stage A tile 128x32 as float4 (8 vec4 per thread)
#pragma unroll
        for (int t = 0; t < 8; ++t) {
            int idx = tid + t * 128;           // 0..1023 vec4 indices
            int m = idx >> 3, c4 = (idx & 7) * 4;
            v4f v = *(const v4f*)&A[(size_t)(m0 + m) * K + k0 + c4];
            *(v4bf*)&As[m][c4] = f2bf4(v);
        }
        // Stage B tile into K-major rows Bs[n][k]
        if (BL) {
#pragma unroll
            for (int t = 0; t < 4; ++t) {
                int idx = tid + t * 128;       // 0..511 vec4
                int n = idx >> 3, c4 = (idx & 7) * 4;
                v4f v = *(const v4f*)&Bw[(size_t)(n0 + n) * K + k0 + c4];
                *(v4bf*)&Bs[n][c4] = f2bf4(v);
            }
        } else {
#pragma unroll
            for (int t = 0; t < 4; ++t) {
                int idx = tid + t * 128;       // 0..511 vec4 along N
                int k = idx >> 4, n4 = (idx & 15) * 4;
                v4f v = *(const v4f*)&Bw[(size_t)(k0 + k) * N + n0 + n4];
#pragma unroll
                for (int j = 0; j < 4; ++j) Bs[n4 + j][k] = f2bf(v[j]);
            }
        }
        __syncthreads();

        const int kb = (lane & 16) ? 8 : 0;
        v16bf a0 = load_frag(&As[wave * 32 + (lane & 15)][0], kb);
        v16bf a1 = load_frag(&As[wave * 32 + 16 + (lane & 15)][0], kb);
#pragma unroll
        for (int t = 0; t < 4; ++t) {
            v16bf b = load_frag(&Bs[t * 16 + (lane & 15)][0], kb);
            acc[0][t] = __builtin_amdgcn_wmma_f32_16x16x32_bf16(
                false, a0, false, b, (short)0, acc[0][t], false, false);
            acc[1][t] = __builtin_amdgcn_wmma_f32_16x16x32_bf16(
                false, a1, false, b, (short)0, acc[1][t], false, false);
        }
        __syncthreads();
    }

    // Epilogue. C/D layout: VGPR r -> M = r (+8 for lanes 16..31), N = lane&15.
#pragma unroll
    for (int h = 0; h < 2; ++h) {
        const int rbase = m0 + wave * 32 + h * 16 + ((lane & 16) ? 8 : 0);
        const int cbase = n0 + (lane & 15);
#pragma unroll
        for (int t = 0; t < 4; ++t) {
            int col = cbase + t * 16;
            float bv = HB ? bias[col] : 0.0f;
#pragma unroll
            for (int r = 0; r < 8; ++r) {
                int row = rbase + r;
                float v = acc[h][t][r] + bv;
                if (HR) v += resid[(size_t)row * N + col];
                if (EPI == 1) v = fmaxf(v, 0.0f);
                else if (EPI == 2) v = gelu_f(v);
                D[(size_t)row * N + col] = v;
            }
        }
    }
}

// ---------------------------------------------------------------------------
// Repack conv weights OIHW [Co,Ci,3,3] -> [tap][Co][Ci] (ci-contiguous).
// ---------------------------------------------------------------------------
__global__ void repack_conv_w_kernel(const float* __restrict__ Wt, float* __restrict__ Wp,
                                     int Cin, int cinShift, long percut) {
    long i = (long)blockIdx.x * 256 + threadIdx.x;   // over Cout*Cin
    if (i >= percut) return;
    int tap = blockIdx.y;
    long co = i >> cinShift;
    int ci = (int)(i & (Cin - 1));
    Wp[(size_t)tap * percut + i] = Wt[((size_t)co * Cin + ci) * 9 + tap];
}

// ---------------------------------------------------------------------------
// Implicit-GEMM 3x3 conv, NHWC seq layout [B,H,W,Cin], repacked weights
// Wp[tap][Co][Ci]. H=W=32. Block: 128 threads (4 waves),
// tile = 16 pixels (half image row) x 64 cout. Zero padding via clamped
// address + select (no divergent branches).
// ---------------------------------------------------------------------------
__global__ void conv3x3_wmma_kernel(const float* __restrict__ X,
                                    const float* __restrict__ Wp,
                                    const float* __restrict__ bias,
                                    float* __restrict__ Y,
                                    int Cin, int Cout, int relu) {
    __shared__ __attribute__((aligned(16))) __bf16 As[16][40];
    __shared__ __attribute__((aligned(16))) __bf16 Bs[64][40];

    const int tid  = threadIdx.x;
    const int wave = tid >> 5;
    const int lane = tid & 31;
    const int pb = blockIdx.x;          // (b, h, half-row)
    const int w0 = (pb & 1) * 16;
    const int h  = (pb >> 1) & 31;
    const int b  = pb >> 6;
    const int co0 = blockIdx.y * 64;
    const size_t percut = (size_t)Cout * Cin;

    v8f acc = (v8f)(0.0f);

    for (int k0 = 0; k0 < Cin; k0 += 32) {
#pragma unroll
        for (int tap = 0; tap < 9; ++tap) {
            const int dy = tap / 3 - 1, dx = tap % 3 - 1;   // compile-time
            const int hh = h + dy;
            const bool hok = (unsigned)hh < 32u;
            const int hcl = min(max(hh, 0), 31);
            // A: 16 pixels x 32 ci, one float4 per thread, clamp+select padding
            {
                int m = tid >> 3, c4 = (tid & 7) * 4;
                int ww = w0 + m + dx;
                bool ok = hok && ((unsigned)ww < 32u);
                int wcl = min(max(ww, 0), 31);
                v4f v = *(const v4f*)&X[(((size_t)b * 32 + hcl) * 32 + wcl) * Cin + k0 + c4];
                if (!ok) v = (v4f)(0.0f);
                *(v4bf*)&As[m][c4] = f2bf4(v);
            }
            // B: 64 cout x 32 ci for this tap (ci-contiguous after repack)
#pragma unroll
            for (int t = 0; t < 4; ++t) {
                int idx = tid + t * 128;       // 0..511 vec4
                int n = idx >> 3, c4 = (idx & 7) * 4;
                v4f v = *(const v4f*)&Wp[(size_t)tap * percut +
                                         (size_t)(co0 + n) * Cin + k0 + c4];
                *(v4bf*)&Bs[n][c4] = f2bf4(v);
            }
            __syncthreads();

            const int kb = (lane & 16) ? 8 : 0;
            v16bf a  = load_frag(&As[lane & 15][0], kb);
            v16bf bb = load_frag(&Bs[wave * 16 + (lane & 15)][0], kb);
            acc = __builtin_amdgcn_wmma_f32_16x16x32_bf16(
                false, a, false, bb, (short)0, acc, false, false);
            __syncthreads();
        }
    }

    const int co = co0 + wave * 16 + (lane & 15);
    const int pw = ((lane & 16) ? 8 : 0);
    const float bv = bias[co];
#pragma unroll
    for (int r = 0; r < 8; ++r) {
        int w = w0 + pw + r;
        float v = acc[r] + bv;
        if (relu) v = fmaxf(v, 0.0f);
        Y[(((size_t)b * 32 + h) * 32 + w) * Cout + co] = v;
    }
}

// ---------------------------------------------------------------------------
// LayerNorm over last dim C=512. One block (128 thr) per row.
// ---------------------------------------------------------------------------
__global__ void layernorm_kernel(const float* __restrict__ X,
                                 const float* __restrict__ g,
                                 const float* __restrict__ bta,
                                 float* __restrict__ Y) {
    __shared__ float red[128];
    const int row = blockIdx.x, tid = threadIdx.x;
    const float* xr = X + (size_t)row * 512;
    float vals[4], s = 0.0f;
#pragma unroll
    for (int i = 0; i < 4; ++i) { vals[i] = xr[tid + i * 128]; s += vals[i]; }
    red[tid] = s; __syncthreads();
    for (int st = 64; st > 0; st >>= 1) { if (tid < st) red[tid] += red[tid + st]; __syncthreads(); }
    const float mean = red[0] * (1.0f / 512.0f);
    __syncthreads();
    float v = 0.0f;
#pragma unroll
    for (int i = 0; i < 4; ++i) { float d = vals[i] - mean; v += d * d; }
    red[tid] = v; __syncthreads();
    for (int st = 64; st > 0; st >>= 1) { if (tid < st) red[tid] += red[tid + st]; __syncthreads(); }
    const float rstd = rsqrtf(red[0] * (1.0f / 512.0f) + 1e-5f);
    float* yr = Y + (size_t)row * 512;
#pragma unroll
    for (int i = 0; i < 4; ++i) {
        int c = tid + i * 128;
        yr[c] = (vals[i] - mean) * rstd * g[c] + bta[c];
    }
}

// ---------------------------------------------------------------------------
// Channel attention: per (b, c=h*64+dd): l2norm q,k over N, softmax(q*k*temp)
// over N, out = softmax * v. qkv laid out [b,n][3*512] (q|k|v).
// ---------------------------------------------------------------------------
__device__ __forceinline__ float blk_sum(float v, float* red) {
    int tid = threadIdx.x;
    __syncthreads(); red[tid] = v; __syncthreads();
    for (int st = 128; st > 0; st >>= 1) { if (tid < st) red[tid] += red[tid + st]; __syncthreads(); }
    return red[0];
}
__device__ __forceinline__ float blk_max(float v, float* red) {
    int tid = threadIdx.x;
    __syncthreads(); red[tid] = v; __syncthreads();
    for (int st = 128; st > 0; st >>= 1) { if (tid < st) red[tid] = fmaxf(red[tid], red[tid + st]); __syncthreads(); }
    return red[0];
}

__global__ void chanattn_kernel(const float* __restrict__ qkv,
                                const float* __restrict__ temp,
                                float* __restrict__ out) {
    __shared__ float red[256];
    const int gc = blockIdx.x;       // b*512 + c
    const int b = gc >> 9, c = gc & 511;
    const float tt = temp[c >> 6];
    const int tid = threadIdx.x;

    float q[4], k[4], v[4], sq = 0.0f, sk = 0.0f;
#pragma unroll
    for (int i = 0; i < 4; ++i) {
        int n = tid + i * 256;
        size_t base = ((size_t)(b * 1024 + n)) * 1536 + c;
        q[i] = qkv[base]; k[i] = qkv[base + 512]; v[i] = qkv[base + 1024];
        sq += q[i] * q[i]; sk += k[i] * k[i];
    }
    sq = blk_sum(sq, red);
    sk = blk_sum(sk, red);
    const float qn = 1.0f / fmaxf(sqrtf(sq), 1e-12f);
    const float kn = 1.0f / fmaxf(sqrtf(sk), 1e-12f);

    float s[4], mx = -3.0e38f;
#pragma unroll
    for (int i = 0; i < 4; ++i) { s[i] = q[i] * qn * k[i] * kn * tt; mx = fmaxf(mx, s[i]); }
    mx = blk_max(mx, red);
    float se = 0.0f;
#pragma unroll
    for (int i = 0; i < 4; ++i) { s[i] = __expf(s[i] - mx); se += s[i]; }
    se = blk_sum(se, red);
    const float inv = 1.0f / se;
#pragma unroll
    for (int i = 0; i < 4; ++i) {
        int n = tid + i * 256;
        out[((size_t)(b * 1024 + n)) * 512 + c] = s[i] * inv * v[i];
    }
}

// ---------------------------------------------------------------------------
// softmax over H (axis=2 of [B,C,H,W]); one thread per (b,w,c).
// ---------------------------------------------------------------------------
__global__ void softmax_h_kernel(float* __restrict__ X) {
    const int g = blockIdx.x * blockDim.x + threadIdx.x;
    if (g >= 16 * 32 * 512) return;
    const int c = g & 511, w = (g >> 9) & 31, b = g >> 14;
    float v[32], mx = -3.0e38f;
#pragma unroll
    for (int h = 0; h < 32; ++h) {
        v[h] = X[((size_t)(b * 1024 + h * 32 + w)) * 512 + c];
        mx = fmaxf(mx, v[h]);
    }
    float s = 0.0f;
#pragma unroll
    for (int h = 0; h < 32; ++h) { v[h] = __expf(v[h] - mx); s += v[h]; }
    const float inv = 1.0f / s;
#pragma unroll
    for (int h = 0; h < 32; ++h)
        X[((size_t)(b * 1024 + h * 32 + w)) * 512 + c] = v[h] * inv;
}

// softmax over channels (C=512), one block (128 thr) per row, in-place.
__global__ void softmax_c_kernel(float* __restrict__ X) {
    __shared__ float red[128];
    const int row = blockIdx.x, tid = threadIdx.x;
    float* xr = X + (size_t)row * 512;
    float v[4], mx = -3.0e38f;
#pragma unroll
    for (int i = 0; i < 4; ++i) { v[i] = xr[tid + i * 128]; mx = fmaxf(mx, v[i]); }
    red[tid] = mx; __syncthreads();
    for (int st = 64; st > 0; st >>= 1) { if (tid < st) red[tid] = fmaxf(red[tid], red[tid + st]); __syncthreads(); }
    mx = red[0]; __syncthreads();
    float s = 0.0f;
#pragma unroll
    for (int i = 0; i < 4; ++i) { v[i] = __expf(v[i] - mx); s += v[i]; }
    red[tid] = s; __syncthreads();
    for (int st = 64; st > 0; st >>= 1) { if (tid < st) red[tid] += red[tid + st]; __syncthreads(); }
    const float inv = 1.0f / red[0];
#pragma unroll
    for (int i = 0; i < 4; ++i) xr[tid + i * 128] = v[i] * inv;
}

// attended = key * value * query * temp[c]
__global__ void eamul_kernel(const float* __restrict__ k, const float* __restrict__ q,
                             const float* __restrict__ v, const float* __restrict__ temp,
                             float* __restrict__ o, long total) {
    long i = (long)blockIdx.x * blockDim.x + threadIdx.x;
    if (i < total) o[i] = k[i] * v[i] * q[i] * temp[i & 511];
}

// SE: column mean over n; [16,1024,512] -> [16,512]
__global__ void colmean_kernel(const float* __restrict__ X, float* __restrict__ s) {
    const int b = blockIdx.x, c = threadIdx.x;
    float a = 0.0f;
    for (int n = 0; n < 1024; ++n) a += X[((size_t)b * 1024 + n) * 512 + c];
    s[b * 512 + c] = a * (1.0f / 1024.0f);
}
// s1[b,r] = gelu(sum_c s[b,c]*rw[r,c] + rb[r]);  16x128 outputs
__global__ void se_fc1_kernel(const float* __restrict__ s, const float* __restrict__ rw,
                              const float* __restrict__ rb, float* __restrict__ s1) {
    int o = blockIdx.x * blockDim.x + threadIdx.x;
    if (o >= 16 * 128) return;
    int b = o >> 7, r = o & 127;
    float a = rb[r];
    for (int c = 0; c < 512; ++c) a += s[b * 512 + c] * rw[r * 512 + c];
    s1[o] = gelu_f(a);
}
// s2[b,c] = sum_r s1[b,r]*ew[c,r] + eb[c];  16x512 outputs
__global__ void se_fc2_kernel(const float* __restrict__ s1, const float* __restrict__ ew,
                              const float* __restrict__ eb, float* __restrict__ s2) {
    int o = blockIdx.x * blockDim.x + threadIdx.x;
    if (o >= 16 * 512) return;
    int b = o >> 9, c = o & 511;
    float a = eb[c];
    for (int r = 0; r < 128; ++r) a += s1[b * 128 + r] * ew[c * 128 + r];
    s2[o] = a;
}
// se residual: y = x * (1 + sigmoid(s2[b,c]))
__global__ void se_scale_kernel(const float* __restrict__ X, const float* __restrict__ s2,
                                float* __restrict__ Y, long total) {
    long i = (long)blockIdx.x * blockDim.x + threadIdx.x;
    if (i >= total) return;
    float g = s2[(i >> 19) * 512 + (i & 511)];
    Y[i] = X[i] * (1.0f + 1.0f / (1.0f + __expf(-g)));
}

// y = relu(x*g[c]+b[c]) with c = i & (Cch-1) (Cch power of two)
__global__ void bnrelu_kernel(const float* __restrict__ X, const float* __restrict__ g,
                              const float* __restrict__ b, float* __restrict__ Y,
                              long total, int cmask) {
    long i = (long)blockIdx.x * blockDim.x + threadIdx.x;
    if (i >= total) return;
    int c = (int)(i & cmask);
    Y[i] = fmaxf(X[i] * g[c] + b[c], 0.0f);
}

// concat along channels: out[row,0:512]=A[row], out[row,512:1024]=B[row]
__global__ void concat_kernel(const float* __restrict__ A, const float* __restrict__ B,
                              float* __restrict__ O, long total) {
    long i = (long)blockIdx.x * blockDim.x + threadIdx.x;
    if (i >= total) return;
    long row = i >> 10; int j = (int)(i & 1023);
    O[i] = (j < 512) ? A[row * 512 + j] : B[row * 512 + (j - 512)];
}

// ---------------------------------------------------------------------------
// Host-side orchestration: compile-time dispatch over used GEMM variants.
// ---------------------------------------------------------------------------
static inline void gemm(hipStream_t st, const float* A, const float* B,
                        const float* bias, const float* resid, float* D,
                        int M, int N, int K, int blayout, int epi) {
    dim3 g(M / 128, N / 64);
#define GLAUNCH(bl, ep, hb, hr) \
    gemm_wmma_kernel<bl, ep, hb, hr><<<g, 128, 0, st>>>(A, B, bias, resid, D, M, N, K)
    if (blayout == 0) {
        if (epi == 2)            GLAUNCH(0, 2, true, false);
        else if (!bias)          GLAUNCH(0, 0, false, false);
        else if (resid)          GLAUNCH(0, 0, true, true);
        else                     GLAUNCH(0, 0, true, false);
    } else {
        if (epi == 1)            GLAUNCH(1, 1, true, false);
        else if (resid)          GLAUNCH(1, 0, true, true);
        else                     GLAUNCH(1, 0, true, false);
    }
#undef GLAUNCH
}

extern "C" void kernel_launch(void* const* d_in, const int* in_sizes, int n_in,
                              void* d_out, int out_size, void* d_ws, size_t ws_size,
                              hipStream_t stream) {
    (void)in_sizes; (void)n_in; (void)out_size; (void)ws_size;
    // d_in order: x, then params flattened in alphabetical pytree order, then H, W.
    const float* x        = (const float*)d_in[0];
    const float* bn1_b    = (const float*)d_in[1];
    const float* bn1_g    = (const float*)d_in[2];
    const float* bn2_b    = (const float*)d_in[3];
    const float* bn2_g    = (const float*)d_in[4];
    const float* bn3_b    = (const float*)d_in[5];
    const float* bn3_g    = (const float*)d_in[6];
    const float* ca_proj_b= (const float*)d_in[7];
    const float* ca_proj_w= (const float*)d_in[8];
    const float* ca_qkv_w = (const float*)d_in[9];
    const float* ca_temp  = (const float*)d_in[10];
    const float* conv1_b  = (const float*)d_in[11];
    const float* conv1_w  = (const float*)d_in[12];
    const float* conv3_b  = (const float*)d_in[13];
    const float* conv3_w  = (const float*)d_in[14];
    const float* ea_k_b   = (const float*)d_in[15];
    const float* ea_k_w   = (const float*)d_in[16];
    const float* ea_q_b   = (const float*)d_in[17];
    const float* ea_q_w   = (const float*)d_in[18];
    const float* ea_r_b   = (const float*)d_in[19];
    const float* ea_r_w   = (const float*)d_in[20];
    const float* ea_temp  = (const float*)d_in[21];
    const float* ea_v_b   = (const float*)d_in[22];
    const float* ea_v_w   = (const float*)d_in[23];
    const float* ln1_b    = (const float*)d_in[24];
    const float* ln1_g    = (const float*)d_in[25];
    const float* ln2_b    = (const float*)d_in[26];
    const float* ln2_g    = (const float*)d_in[27];
    // d_in[28],[29]: ln3 (unused by forward)
    const float* ln4_b    = (const float*)d_in[30];
    const float* ln4_g    = (const float*)d_in[31];
    const float* ln5_b    = (const float*)d_in[32];
    const float* ln5_g    = (const float*)d_in[33];
    const float* ln6_b    = (const float*)d_in[34];
    const float* ln6_g    = (const float*)d_in[35];
    const float* m1_b1    = (const float*)d_in[36];
    const float* m1_b2    = (const float*)d_in[37];
    const float* m1_w1    = (const float*)d_in[38];
    const float* m1_w2    = (const float*)d_in[39];
    const float* m2_b1    = (const float*)d_in[40];
    const float* m2_b2    = (const float*)d_in[41];
    const float* m2_w1    = (const float*)d_in[42];
    const float* m2_w2    = (const float*)d_in[43];
    const float* r1_b     = (const float*)d_in[44];
    const float* r1_w     = (const float*)d_in[45];
    const float* r2_b     = (const float*)d_in[46];
    const float* r2_w     = (const float*)d_in[47];
    const float* r3_b     = (const float*)d_in[48];
    const float* r3_w     = (const float*)d_in[49];
    const float* se_eb    = (const float*)d_in[50];
    const float* se_ew    = (const float*)d_in[51];
    const float* se_rb    = (const float*)d_in[52];
    const float* se_rw    = (const float*)d_in[53];

    const int M = 16 * 1024;                 // B*N rows
    const size_t S = (size_t)M * 512;        // 8M floats
    float* ws = (float*)d_ws;
    float* R0 = ws;            // S
    float* R1 = ws + S;        // S
    float* R2 = ws + 2 * S;    // S
    float* R3 = ws + 3 * S;    // S
    float* R4 = ws + 4 * S;    // 4S (qkv / MLP hidden / concat)
    float* R5 = ws + 8 * S;    // S  (mlp2 branch output, long-lived)
    float* sm0 = ws + 9 * S;            // 16*512
    float* sm1 = sm0 + 16 * 512;        // 16*128
    float* sm2 = sm1 + 16 * 128;        // 16*512
    float* wp1 = sm2 + 16 * 512;        // repacked conv3_w: 9*512*512
    float* wp2 = wp1 + (size_t)9 * 512 * 512;  // repacked r2_w: 9*256*256

    const long SL = (long)S;
    const int EB = (int)((S + 255) / 256);

    // ---- one-shot weight repack for the two 3x3 convs ----
    repack_conv_w_kernel<<<dim3((512 * 512) / 256, 9), 256, 0, stream>>>(
        conv3_w, wp1, 512, 9, (long)512 * 512);
    repack_conv_w_kernel<<<dim3((256 * 256) / 256, 9), 256, 0, stream>>>(
        r2_w, wp2, 256, 8, (long)256 * 256);

    // ---- channel-attention branch ----
    layernorm_kernel<<<M, 128, 0, stream>>>(x, ln5_g, ln5_b, R0);             // seg
    gemm(stream, R0, ca_qkv_w, nullptr, nullptr, R4, M, 1536, 512, 0, 0);     // qkv
    chanattn_kernel<<<16 * 512, 256, 0, stream>>>(R4, ca_temp, R1);           // ca out
    gemm(stream, R1, ca_proj_w, ca_proj_b, R0, R2, M, 512, 512, 0, 0);        // add3
    layernorm_kernel<<<M, 128, 0, stream>>>(R2, ln4_g, ln4_b, R3);            // n4
    colmean_kernel<<<16, 512, 0, stream>>>(R3, sm0);
    se_fc1_kernel<<<8, 256, 0, stream>>>(sm0, se_rw, se_rb, sm1);
    se_fc2_kernel<<<32, 256, 0, stream>>>(sm1, se_ew, se_eb, sm2);
    se_scale_kernel<<<EB, 256, 0, stream>>>(R3, sm2, R0, SL);                 // se (pre-LN)
    layernorm_kernel<<<M, 128, 0, stream>>>(R0, ln6_g, ln6_b, R1);            // se (post-LN)
    gemm(stream, R1, m2_w1, m2_b1, nullptr, R4, M, 2048, 512, 0, 2);          // gelu fc1
    gemm(stream, R4, m2_w2, m2_b2, R1, R5, M, 512, 2048, 0, 0);               // mlp2 out

    // ---- spatial (efficient attention) branch ----
    layernorm_kernel<<<M, 128, 0, stream>>>(x, ln1_g, ln1_b, R0);             // n1
    conv3x3_wmma_kernel<<<dim3(1024, 8), 128, 0, stream>>>(R0, wp1, conv3_b, R1, 512, 512, 1);
    gemm(stream, R1, ea_k_w, ea_k_b, nullptr, R2, M, 512, 512, 1, 0);         // keys
    gemm(stream, R1, ea_q_w, ea_q_b, nullptr, R3, M, 512, 512, 1, 0);         // queries
    gemm(stream, R1, ea_v_w, ea_v_b, nullptr, R0, M, 512, 512, 1, 0);         // values
    softmax_h_kernel<<<1024, 256, 0, stream>>>(R2);                           // key softmax over H
    softmax_c_kernel<<<M, 128, 0, stream>>>(R3);                              // query softmax over C
    eamul_kernel<<<EB, 256, 0, stream>>>(R2, R3, R0, ea_temp, R1, SL);        // attended
    gemm(stream, R1, ea_r_w, ea_r_b, x, R2, M, 512, 512, 1, 0);               // add1 = x + attn
    layernorm_kernel<<<M, 128, 0, stream>>>(R2, ln2_g, ln2_b, R0);
    gemm(stream, R0, m1_w1, m1_b1, nullptr, R4, M, 2048, 512, 0, 2);          // gelu fc1
    gemm(stream, R4, m1_w2, m1_b2, R2, R3, M, 512, 2048, 0, 0);               // add2

    // ---- merge + residual bottleneck ----
    concat_kernel<<<(int)((2 * S + 255) / 256), 256, 0, stream>>>(R3, R5, R4, (long)(2 * S));
    gemm(stream, R4, conv1_w, conv1_b, nullptr, R0, M, 512, 1024, 1, 1);      // mi = relu(...)
    bnrelu_kernel<<<EB, 256, 0, stream>>>(R0, bn1_g, bn1_b, R1, SL, 511);
    gemm(stream, R1, r1_w, r1_b, nullptr, R2, M, 256, 512, 1, 0);             // h1 [M,256]
    bnrelu_kernel<<<EB / 2, 256, 0, stream>>>(R2, bn2_g, bn2_b, R1, SL / 2, 255);
    conv3x3_wmma_kernel<<<dim3(1024, 4), 128, 0, stream>>>(R1, wp2, r2_b, R3, 256, 256, 0);
    bnrelu_kernel<<<EB / 2, 256, 0, stream>>>(R3, bn3_g, bn3_b, R1, SL / 2, 255);
    gemm(stream, R1, r3_w, r3_b, R0, (float*)d_out, M, 512, 256, 1, 0);       // mx -> out
}
